// DynamicSparseAttention_3298534884132
// MI455X (gfx1250) — matrix operand order repaired and checked by reading, not compiled, not run on hardware
//
#include <hip/hip_runtime.h>

// ---------------------------------------------------------------------------
// DynamicSparseAttention on MI455X (gfx1250): bf16 WMMA + TDM/async-LDS moves.
// B=2, S=2048, D=2048, H=16, hd=128, WINDOW=32, top-k = S/4 = 512.
// ---------------------------------------------------------------------------

typedef __attribute__((ext_vector_type(16))) __bf16 v16bf;
typedef __attribute__((ext_vector_type(8)))  float  v8f;
typedef __attribute__((ext_vector_type(4)))  unsigned u32x4;
typedef __attribute__((ext_vector_type(8)))  int      i32x8;
typedef __attribute__((ext_vector_type(4)))  int      i32x4;

union BF16x16 { v16bf v; unsigned u[8]; };

#define S_LEN 2048
#define D_MOD 2048
#define NH    16
#define HD    128
#define WIN   32
#define KSEL  512
#define SCP   2052   // score-row pitch in f32 (bank-conflict-friendly)

#define AS1 __attribute__((address_space(1)))
#define AS3 __attribute__((address_space(3)))

// CDNA5 data-mover feature detection (device pass only; silent fallback).
#if defined(__HIP_DEVICE_COMPILE__) && defined(__gfx1250__)
  #if __has_builtin(__builtin_amdgcn_global_load_async_to_lds_b128) && \
      __has_builtin(__builtin_amdgcn_s_wait_asynccnt)
    #define HAVE_ASYNC_LDS 1
  #endif
  #if __has_builtin(__builtin_amdgcn_tensor_load_to_lds) && \
      __has_builtin(__builtin_amdgcn_s_wait_tensorcnt)
    #define HAVE_TDM 1
  #endif
#endif

// ---------------- small helpers ----------------
__device__ inline unsigned short f32_to_bf16(float f) {
  unsigned u = __float_as_uint(f);
  unsigned r = u + 0x7fffu + ((u >> 16) & 1u);   // RNE
  return (unsigned short)(r >> 16);
}
__device__ inline unsigned pack_bf16x2(float lo, float hi) {
  return (unsigned)f32_to_bf16(lo) | ((unsigned)f32_to_bf16(hi) << 16);
}
// order-preserving f32 <-> u32 (for radix threshold select)
__device__ inline unsigned f2ord(float f) {
  unsigned u = __float_as_uint(f);
  return (u & 0x80000000u) ? ~u : (u | 0x80000000u);
}
__device__ inline float ord2f(unsigned o) {
  unsigned u = (o & 0x80000000u) ? (o & 0x7fffffffu) : ~o;
  return __uint_as_float(u);
}
__device__ inline int wave_isum(int v) {
  #pragma unroll
  for (int o = 16; o; o >>= 1) v += __shfl_xor(v, o, 32);
  return v;
}
__device__ inline float wave_fsum(float v) {
  #pragma unroll
  for (int o = 16; o; o >>= 1) v += __shfl_xor(v, o, 32);
  return v;
}
__device__ inline float wave_fmax(float v) {
  #pragma unroll
  for (int o = 16; o; o >>= 1) { float t = __shfl_xor(v, o, 32); v = t > v ? t : v; }
  return v;
}

// ---------------- cast fp32 -> bf16 ----------------
__global__ void cast_f32_to_bf16(const float* __restrict__ in,
                                 unsigned short* __restrict__ out, int n) {
  int i = blockIdx.x * blockDim.x + threadIdx.x;
  if (i < n) out[i] = f32_to_bf16(in[i]);
}

// ---------------- tiled bf16 WMMA GEMM ----------------
// C[M,N] = A[M,K] * B[K,N]; A,B bf16 row-major.
// MODE 0: store bf16 row-major. MODE 1: store f32 row-major.
// MODE 2: store bf16 as Vt[b,h,d,s] (per-head transposed), M=B*S, N=H*hd.
#define GTM 128
#define GTN 64
#define GTK 32

template <int MODE>
__global__ __launch_bounds__(256)
void gemm_bf16_wmma(const unsigned short* __restrict__ A,
                    const unsigned short* __restrict__ Bw,
                    void* __restrict__ Cout, int M, int N, int K) {
  __shared__ unsigned As[GTM * (GTK / 2)];    // 128 rows x 16 u32 (32 bf16 K)
  __shared__ unsigned BsT[GTN * (GTK / 2)];   // 64 cols x 16 u32 (K-major)

  const int tid  = threadIdx.x;
  const int lane = tid & 31;
  const int wv   = tid >> 5;
  const int wy   = wv >> 1, wx = wv & 1;
  const int half = lane >> 4;
  const int l15  = lane & 15;
  const int m0   = blockIdx.y * GTM;
  const int n0   = blockIdx.x * GTN;

  const unsigned* Ag = (const unsigned*)A;
  const unsigned* Bg = (const unsigned*)Bw;
  const int Ku = K >> 1, Nu = N >> 1;

  v8f acc[2][2];
  #pragma unroll
  for (int i = 0; i < 2; ++i)
    #pragma unroll
    for (int j = 0; j < 2; ++j)
      #pragma unroll
      for (int r = 0; r < 8; ++r) acc[i][j][r] = 0.f;

  for (int k0 = 0; k0 < K; k0 += GTK) {
#ifdef HAVE_ASYNC_LDS
    // stage A tile with the async-LDS engine: 512 x b128, 2 per thread
    #pragma unroll
    for (int i = 0; i < 2; ++i) {
      int e = tid + i * 256;                  // b128 id 0..511
      int r = e >> 2, c4 = (e & 3) * 4;       // u32 col 0/4/8/12
      const unsigned* gp = &Ag[(size_t)(m0 + r) * Ku + (k0 >> 1) + c4];
      __builtin_amdgcn_global_load_async_to_lds_b128(
          (AS1 i32x4*)gp, (AS3 i32x4*)&As[e * 4], 0, 0);
    }
#else
    #pragma unroll
    for (int i = 0; i < 8; ++i) {
      int e = tid + i * 256;
      int r = e >> 4, c = e & 15;
      As[e] = Ag[(size_t)(m0 + r) * Ku + (k0 >> 1) + c];
    }
#endif
    // stage B tile transposed to K-major: 1024 u32 reads, u16 scatter
    unsigned short* BsT16 = (unsigned short*)BsT;
    #pragma unroll
    for (int i = 0; i < 4; ++i) {
      int e = tid + i * 256;
      int kr = e >> 5, c = e & 31;
      unsigned val = Bg[(size_t)(k0 + kr) * Nu + (n0 >> 1) + c];
      BsT16[(2 * c) * GTK + kr]     = (unsigned short)(val & 0xffffu);
      BsT16[(2 * c + 1) * GTK + kr] = (unsigned short)(val >> 16);
    }
    if (k0 + GTK < K) {  // warm L2 for next K-chunk (global_prefetch_b8)
      __builtin_prefetch(&Ag[(size_t)(m0 + (tid >> 1)) * Ku + ((k0 + GTK) >> 1) + (tid & 1) * 8], 0, 1);
      __builtin_prefetch(&Bg[(size_t)(k0 + GTK + (tid >> 3)) * Nu + (n0 >> 1) + (tid & 7) * 4], 0, 1);
    }
#ifdef HAVE_ASYNC_LDS
    __builtin_amdgcn_s_wait_asynccnt(0);
#endif
    __syncthreads();

    BF16x16 afr[2], bfr[2];
    #pragma unroll
    for (int mt = 0; mt < 2; ++mt) {
      int row = wy * 32 + mt * 16 + l15;
      #pragma unroll
      for (int vv = 0; vv < 8; ++vv) {
        int j = half * 4 + (vv < 4 ? vv : 8 + (vv - 4));  // ISA A-frag K map
        afr[mt].u[vv] = As[row * 16 + j];
      }
    }
    #pragma unroll
    for (int nt = 0; nt < 2; ++nt) {
      int col = wx * 32 + nt * 16 + l15;
      #pragma unroll
      for (int vv = 0; vv < 8; ++vv)
        bfr[nt].u[vv] = BsT[col * 16 + half * 4 + (vv < 4 ? vv : 8 + (vv - 4))];     // ISA table K map
    }
    #pragma unroll
    for (int mt = 0; mt < 2; ++mt)
      #pragma unroll
      for (int nt = 0; nt < 2; ++nt)
        acc[mt][nt] = __builtin_amdgcn_wmma_f32_16x16x32_bf16(
            false, afr[mt].v, false, bfr[nt].v, (short)0, acc[mt][nt], false, false);
    __syncthreads();
  }

  #pragma unroll
  for (int mt = 0; mt < 2; ++mt)
    #pragma unroll
    for (int nt = 0; nt < 2; ++nt)
      #pragma unroll
      for (int r = 0; r < 8; ++r) {
        int row = m0 + wy * 32 + mt * 16 + r + half * 8;
        int col = n0 + wx * 32 + nt * 16 + l15;
        float v = acc[mt][nt][r];
        if (MODE == 1) {
          ((float*)Cout)[(size_t)row * N + col] = v;
        } else if (MODE == 0) {
          ((unsigned short*)Cout)[(size_t)row * N + col] = f32_to_bf16(v);
        } else {  // Vt[b,h,d,s]
          int b = row >> 11, s = row & 2047;
          int h = col >> 7,  d = col & 127;
          ((unsigned short*)Cout)[(((size_t)(b * NH + h) * HD + d) << 11) + s] =
              f32_to_bf16(v);
        }
      }
}

// ---------------- sparse attention core ----------------
// grid (S/32, H, B), block 1024 (32 waves). Dynamic LDS:
//   sc[32][SCP] f32 scores/probs (262656 B) + qs[32][64] u32 Q tile (8192 B).
__global__ __launch_bounds__(1024)
void attn_sparse_wmma(const unsigned short* __restrict__ Qb,
                      const unsigned short* __restrict__ Kb,
                      const unsigned short* __restrict__ Vt,
                      unsigned short* __restrict__ Ob) {
  extern __shared__ char smem[];
  float*    sc = (float*)smem;
  unsigned* qs = (unsigned*)(smem + (size_t)32 * SCP * sizeof(float));

  const int tid  = threadIdx.x;
  const int lane = tid & 31;
  const int w    = tid >> 5;
  const int half = lane >> 4;
  const int l15  = lane & 15;
  const int h    = blockIdx.y;
  const int b    = blockIdx.z;
  const int qbase = blockIdx.x * 32;
  const float scale = 0.08838834764831845f;  // 1/sqrt(128)

  const unsigned* Qg = (const unsigned*)Qb;
  const unsigned* Kg = (const unsigned*)Kb;
  const unsigned* Vg = (const unsigned*)Vt;

#ifdef HAVE_TDM
  // Tensor Data Mover: 2-D tile 128 elems x 32 rows of bf16, stride 2048,
  // from Q[b, qbase.., h*HD..] into qs (LDS byte offset 32*SCP*4; no static LDS).
  if (w == 0) {
    unsigned long long ga =
        (unsigned long long)(const void*)(Qb + ((size_t)(b * S_LEN + qbase) * D_MOD + h * HD));
    const unsigned lds_off = 32u * SCP * 4u;
    u32x4 g0;
    g0[0] = 1u;                                   // count=1, user descriptor
    g0[1] = lds_off;                              // lds_addr
    g0[2] = (unsigned)ga;                         // global_addr[31:0]
    g0[3] = (unsigned)((ga >> 32) & 0x01ffffffull) | (2u << 30);  // addr[56:32] | type=2
    const unsigned td0 = D_MOD, td1 = 2 * S_LEN;  // tensor dims (elements)
    const unsigned tl0 = HD, tl1 = 32;            // tile dims
    const unsigned st0 = D_MOD;                   // dim0 stride (elements)
    i32x8 g1;
    g1[0] = (int)(1u << 16);                      // data_size=2B; no mask/pad/iter
    g1[1] = (int)((td0 & 0xffffu) << 16);         // tensor_dim0[15:0] @ bit48
    g1[2] = (int)((td0 >> 16) | ((td1 & 0xffffu) << 16));
    g1[3] = (int)((td1 >> 16) | (tl0 << 16));     // tile_dim0 @ bit112
    g1[4] = (int)(tl1);                           // tile_dim1; tile_dim2=0
    g1[5] = (int)st0;                             // tensor_dim0_stride[31:0]
    g1[6] = 0;                                    // stride0[47:32] | stride1[15:0]
    g1[7] = 0;                                    // stride1[47:16]
    i32x4 gz = {0, 0, 0, 0};
#if defined(__clang_major__) && (__clang_major__ >= 23)
    i32x8 gz8 = {0, 0, 0, 0, 0, 0, 0, 0};
    __builtin_amdgcn_tensor_load_to_lds(g0, g1, gz, gz, gz8, 0);
#else
    __builtin_amdgcn_tensor_load_to_lds(g0, g1, gz, gz, 0);
#endif
    __builtin_amdgcn_s_wait_tensorcnt(0);
  }
#else
  // stage Q tile (32 x 128 bf16 = 2048 u32)
  #pragma unroll
  for (int t = 0; t < 2; ++t) {
    int e = tid + t * 1024;
    int r = e >> 6, c = e & 63;
    qs[e] = Qg[(size_t)(b * S_LEN + qbase + r) * (D_MOD / 2) + h * (HD / 2) + c];
  }
#endif
  (void)Qg;
  __syncthreads();

  // ---- Phase 1: scores = (Q K^T) * scale ; 256 16x16 tiles / 32 waves ----
  #pragma unroll 1
  for (int t = 0; t < 8; ++t) {
    int tileid = w + t * 32;
    int qt = tileid >> 7;       // 0..1
    int kt = tileid & 127;      // 0..127
    v8f acc;
    #pragma unroll
    for (int r = 0; r < 8; ++r) acc[r] = 0.f;
    int qrow = qt * 16 + l15;
    int key  = kt * 16 + l15;
    const unsigned* Krow = Kg + (size_t)(b * S_LEN + key) * (D_MOD / 2) + h * (HD / 2);
    #pragma unroll
    for (int kk = 0; kk < HD; kk += 32) {
      BF16x16 a, bb;
      #pragma unroll
      for (int vv = 0; vv < 8; ++vv) {
        int j = half * 4 + (vv < 4 ? vv : 8 + (vv - 4));
        a.u[vv]  = qs[qrow * 64 + (kk >> 1) + j];
        bb.u[vv] = Krow[(kk >> 1) + j];
      }
      acc = __builtin_amdgcn_wmma_f32_16x16x32_bf16(
          false, a.v, false, bb.v, (short)0, acc, false, false);
    }
    #pragma unroll
    for (int r = 0; r < 8; ++r) {
      int row = qt * 16 + r + half * 8;
      int col = kt * 16 + l15;
      sc[row * SCP + col] = acc[r] * scale;
    }
  }
  __syncthreads();

  // ---- Phase 2+3: per-row top-512 threshold (radix select), mask, softmax --
  {
    unsigned uu[64];
    #pragma unroll
    for (int i = 0; i < 64; ++i)
      uu[i] = f2ord(sc[w * SCP + i * 32 + lane]);

    unsigned cur = 0u;
    for (int bit = 31; bit >= 0; --bit) {
      unsigned cand = cur | (1u << bit);
      int cnt = 0;
      #pragma unroll
      for (int i = 0; i < 64; ++i) cnt += (uu[i] >= cand) ? 1 : 0;
      cnt = wave_isum(cnt);
      if (cnt >= KSEL) cur = cand;
    }

    const int q = qbase + w;
    float m = -3.0e38f;
    #pragma unroll
    for (int i = 0; i < 64; ++i) {
      int col = i * 32 + lane;
      int dd = col - q; if (dd < 0) dd = -dd;
      bool keep = (uu[i] >= cur) || (dd <= WIN);
      float f = ord2f(uu[i]);
      if (keep && f > m) m = f;
    }
    m = wave_fmax(m);

    float ssum = 0.f;
    #pragma unroll
    for (int i = 0; i < 64; ++i) {
      int col = i * 32 + lane;
      int dd = col - q; if (dd < 0) dd = -dd;
      bool keep = (uu[i] >= cur) || (dd <= WIN);
      float ev = keep ? __expf(ord2f(uu[i]) - m) : 0.f;
      ssum += ev;
      sc[w * SCP + col] = ev;
    }
    ssum = wave_fsum(ssum);
    float inv = 1.f / ssum;
    #pragma unroll
    for (int i = 0; i < 64; ++i)
      sc[w * SCP + i * 32 + lane] *= inv;
  }
  __syncthreads();

  // ---- Phase 4: out = probs @ V  (16 output tiles, waves 0..15) ----
  if (w < 16) {
    int qt = w >> 3;   // 0..1
    int dt = w & 7;    // 0..7 (16-wide slices of hd)
    v8f acc;
    #pragma unroll
    for (int r = 0; r < 8; ++r) acc[r] = 0.f;
    int arow = qt * 16 + l15;
    const unsigned* Vcol =
        Vg + (((size_t)(b * NH + h) * HD + dt * 16 + l15) << 11) / 2;  // Vt u32
    #pragma unroll 1
    for (int kc = 0; kc < S_LEN; kc += 32) {
      BF16x16 a, bb;
      #pragma unroll
      for (int vv = 0; vv < 8; ++vv) {
        int koff = half * 8 + (vv < 4 ? 2 * vv : 16 + 2 * (vv - 4));
        float f0 = sc[arow * SCP + kc + koff];
        float f1 = sc[arow * SCP + kc + koff + 1];
        a.u[vv]  = pack_bf16x2(f0, f1);
        bb.u[vv] = Vcol[((kc + koff) >> 1)];
      }
      acc = __builtin_amdgcn_wmma_f32_16x16x32_bf16(
          false, a.v, false, bb.v, (short)0, acc, false, false);
    }
    #pragma unroll
    for (int r = 0; r < 8; ++r) {
      int srow = qbase + qt * 16 + r + half * 8;
      int col  = h * HD + dt * 16 + l15;
      Ob[(size_t)(b * S_LEN + srow) * D_MOD + col] = f32_to_bf16(acc[r]);
    }
  }
}

// ---------------- host orchestration ----------------
extern "C" void kernel_launch(void* const* d_in, const int* in_sizes, int n_in,
                              void* d_out, int out_size, void* d_ws, size_t ws_size,
                              hipStream_t stream) {
  (void)in_sizes; (void)n_in; (void)out_size; (void)ws_size;
  const float* hs = (const float*)d_in[0];
  const float* Wq = (const float*)d_in[1];
  const float* Wk = (const float*)d_in[2];
  const float* Wv = (const float*)d_in[3];
  const float* Wo = (const float*)d_in[4];

  const size_t nHS = (size_t)2 * S_LEN * D_MOD;   // 8,388,608
  const size_t nW  = (size_t)D_MOD * D_MOD;       // 4,194,304
  char* ws = (char*)d_ws;
  unsigned short* hsb = (unsigned short*)ws;                      ws += nHS * 2;
  unsigned short* wqb = (unsigned short*)ws;                      ws += nW * 2;
  unsigned short* wkb = (unsigned short*)ws;                      ws += nW * 2;
  unsigned short* wvb = (unsigned short*)ws;                      ws += nW * 2;
  unsigned short* wob = (unsigned short*)ws;                      ws += nW * 2;
  unsigned short* Qb  = (unsigned short*)ws;                      ws += nHS * 2;
  unsigned short* Kb  = (unsigned short*)ws;                      ws += nHS * 2;
  unsigned short* Vt  = (unsigned short*)ws;                      ws += nHS * 2;
  unsigned short* Ob  = (unsigned short*)ws;                      ws += nHS * 2;

  // 1) casts
  cast_f32_to_bf16<<<(int)(nHS / 256), 256, 0, stream>>>(hs, hsb, (int)nHS);
  cast_f32_to_bf16<<<(int)(nW / 256), 256, 0, stream>>>(Wq, wqb, (int)nW);
  cast_f32_to_bf16<<<(int)(nW / 256), 256, 0, stream>>>(Wk, wkb, (int)nW);
  cast_f32_to_bf16<<<(int)(nW / 256), 256, 0, stream>>>(Wv, wvb, (int)nW);
  cast_f32_to_bf16<<<(int)(nW / 256), 256, 0, stream>>>(Wo, wob, (int)nW);

  // 2) projections: M=4096, N=2048, K=2048
  const int M = 2 * S_LEN, N = D_MOD, K = D_MOD;
  dim3 ggrid(N / GTN, M / GTM);  // (32, 32)
  gemm_bf16_wmma<0><<<ggrid, 256, 0, stream>>>(hsb, wqb, Qb, M, N, K);
  gemm_bf16_wmma<0><<<ggrid, 256, 0, stream>>>(hsb, wkb, Kb, M, N, K);
  gemm_bf16_wmma<2><<<ggrid, 256, 0, stream>>>(hsb, wvb, Vt, M, N, K);

  // 3) attention (dynamic LDS: 32*SCP*4 + 32*64*4 = 270,848 B < 320 KB/WGP)
  size_t lds_bytes = (size_t)32 * SCP * sizeof(float) + (size_t)32 * 64 * 4;
  dim3 agrid(S_LEN / 32, NH, 2);
  attn_sparse_wmma<<<agrid, 1024, lds_bytes, stream>>>(Qb, Kb, Vt, Ob);

  // 4) output projection -> f32 d_out
  gemm_bf16_wmma<1><<<ggrid, 256, 0, stream>>>(Ob, wob, (float*)d_out, M, N, K);
}
